// ProductGraphsModel_39479339385251
// MI455X (gfx1250) — matrix operand order repaired and checked by reading, not compiled, not run on hardware
//
#include <hip/hip_runtime.h>
#include <hip/hip_bf16.h>
#include <stdint.h>

#define G_   256
#define N_   512
#define E_   16384
#define H_   256
#define FIN_ 4
#define FG_  6
#define C_   3
#define L_   5

typedef __attribute__((ext_vector_type(16))) __bf16       v16bf;
typedef __attribute__((ext_vector_type(8)))  float        v8f;
typedef __attribute__((ext_vector_type(4)))  unsigned int u32x4;
typedef __attribute__((ext_vector_type(4)))  int          v4i;

typedef __attribute__((address_space(1))) v4i* gv4i_p;    // global int4*
typedef __attribute__((address_space(3))) v4i* lv4i_p;    // LDS int4*

union FragBF { u32x4 q[2]; v16bf v; };

#if __has_builtin(__builtin_amdgcn_global_load_async_to_lds_b128) && \
    __has_builtin(__builtin_amdgcn_s_wait_asynccnt)
#define USE_ASYNC_STAGE 1
#else
#define USE_ASYNC_STAGE 0
#endif

__device__ __forceinline__ unsigned short f2bf(float f) {
  unsigned int u = __float_as_uint(f);
  u += 0x7FFFu + ((u >> 16) & 1u);          // round-to-nearest-even
  return (unsigned short)(u >> 16);
}
__device__ __forceinline__ float bf2f(unsigned short s) {
  return __uint_as_float(((unsigned int)s) << 16);
}

// ---------------- degree / norm precompute ----------------
__global__ void deg_init(float* __restrict__ d) {
  int i = blockIdx.x * blockDim.x + threadIdx.x;      // G*N threads
  d[i] = 1.0f;                                        // self-loop weight
}

__global__ void deg_accum(const int* __restrict__ ei, const float* __restrict__ ew,
                          float* __restrict__ deg) {
  int idx = blockIdx.x * blockDim.x + threadIdx.x;    // G*E threads
  int g = idx / E_;
  int e = idx - g * E_;
  int t = ei[(size_t)g * 2 * E_ + E_ + e];            // dst row of edge_index
  atomicAdd(&deg[g * N_ + t], ew[(size_t)g * E_ + e]);
}

__global__ void deg_rsqrt(float* __restrict__ d) {
  int i = blockIdx.x * blockDim.x + threadIdx.x;      // G*N threads
  float v = d[i];
  d[i] = (v > 0.0f) ? rsqrtf(v) : 0.0f;
}

// ---------------- layer 0: x[GN,4] @ W0[4,256] (f32) ----------------
__global__ void l0_matmul(const float* __restrict__ xin, const float* __restrict__ W0,
                          float* __restrict__ h) {
  size_t idx = (size_t)blockIdx.x * blockDim.x + threadIdx.x;  // G*N*H threads
  size_t gn = idx >> 8;
  int c = (int)(idx & (H_ - 1));
  const float* xr = xin + gn * FIN_;
  float s = 0.0f;
#pragma unroll
  for (int j = 0; j < FIN_; ++j) s += xr[j] * W0[j * H_ + c];
  h[idx] = s;
}

// ---------------- weight transpose + bf16 convert: Wt[l][n][k] = Wh[l][k][n] ----------------
__global__ void wt_conv(const float* __restrict__ Wh, unsigned short* __restrict__ Wt) {
  int idx = blockIdx.x * blockDim.x + threadIdx.x;    // (L-1)*H*H threads
  int l = idx / (H_ * H_);
  int r = idx - l * (H_ * H_);
  int n = r / H_;
  int k = r - n * H_;
  Wt[idx] = f2bf(Wh[(size_t)l * H_ * H_ + (size_t)k * H_ + n]);
}

// ---------------- WMMA GEMM with LDS-staged B panel ----------------
// Block (8 waves) computes 256(M) x 64(N): each wave owns 32(M) x 64(N)
// (2 m-tiles x 4 n-tiles, 8 f32 accumulators). B panel Wt[n0..n0+63][0..255]
// async-staged once into LDS (padded rows: 264 shorts -> conflict-free b128
// reads). A (global) and B (LDS) fragments are software-pipelined one k-step
// ahead so WMMAs overlap the loads.
#define BPAD   264                                    // 256 + 8 shorts padding
#define BROWS  64
__global__ void __launch_bounds__(256)
gemm_bf16_wmma(const unsigned short* __restrict__ X,
               const unsigned short* __restrict__ Wt,
               float* __restrict__ Y, int rows) {
  __shared__ unsigned short Bs[BROWS * BPAD];         // ~33 KB

  const int nSuper = H_ / 64;                         // 4 column strips
  int n0 = (blockIdx.x % nSuper) * 64;
  int m0 = (blockIdx.x / nSuper) * 256 + (threadIdx.x >> 5) * 32;
  int lane = threadIdx.x & 31;

  // ---- stage B panel: 64 rows x 256 shorts, 16B chunks ----
  {
    const unsigned short* src = Wt + (size_t)n0 * H_;
    for (int i = threadIdx.x; i < (BROWS * H_) / 8; i += 256) {
      int r  = i >> 5;                                // row (32 chunks of 8 shorts per row)
      int cc = i & 31;
#if USE_ASYNC_STAGE
      __builtin_amdgcn_global_load_async_to_lds_b128(
          (gv4i_p)(src + (size_t)r * H_ + cc * 8),
          (lv4i_p)(Bs + (size_t)r * BPAD + cc * 8),
          0, 0);
#else
      *(u32x4*)(Bs + (size_t)r * BPAD + cc * 8) =
          *(const u32x4*)(src + (size_t)r * H_ + cc * 8);
#endif
    }
#if USE_ASYNC_STAGE
    __builtin_amdgcn_s_wait_asynccnt(0);
#endif
    __syncthreads();
  }

  if (m0 >= rows) return;

  int lm = lane & 15;
  int ah = (lane < 16) ? 0 : 8;                       // A K-half (ISA 16-bit A 16x32 layout)
  int bh = (lane < 16) ? 0 : 8;                       // B K-half (ISA 16-bit B 32x16 layout)
  const unsigned short* arow0 = X + (size_t)(m0 + lm) * H_ + ah;
  const unsigned short* arow1 = arow0 + (size_t)16 * H_;
  const unsigned short* bbase = Bs + (size_t)lm * BPAD + bh;

  v8f c[8] = {};
  FragBF a0, a1, b[4], an0, an1, bn[4];

  // prologue: fragments for k = 0
  a0.q[0] = *(const u32x4*)(arow0);
  a0.q[1] = *(const u32x4*)(arow0 + 16);
  a1.q[0] = *(const u32x4*)(arow1);
  a1.q[1] = *(const u32x4*)(arow1 + 16);
#pragma unroll
  for (int t = 0; t < 4; ++t) {
    const unsigned short* bp = bbase + (size_t)(t * 16) * BPAD;
    b[t].q[0] = *(const u32x4*)(bp);
    b[t].q[1] = *(const u32x4*)(bp + 16);
  }

#pragma unroll
  for (int ks = 0; ks < 8; ++ks) {
    if (ks < 7) {                                     // prefetch k-step ks+1
      int kn = (ks + 1) * 32;
      an0.q[0] = *(const u32x4*)(arow0 + kn);
      an0.q[1] = *(const u32x4*)(arow0 + kn + 16);
      an1.q[0] = *(const u32x4*)(arow1 + kn);
      an1.q[1] = *(const u32x4*)(arow1 + kn + 16);
#pragma unroll
      for (int t = 0; t < 4; ++t) {
        const unsigned short* bp = bbase + (size_t)(t * 16) * BPAD + kn;
        bn[t].q[0] = *(const u32x4*)(bp);
        bn[t].q[1] = *(const u32x4*)(bp + 16);
      }
    }
#pragma unroll
    for (int t = 0; t < 4; ++t) {
      c[t] = __builtin_amdgcn_wmma_f32_16x16x32_bf16(
          false, a0.v, false, b[t].v, (short)0, c[t], false, false);
      c[4 + t] = __builtin_amdgcn_wmma_f32_16x16x32_bf16(
          false, a1.v, false, b[t].v, (short)0, c[4 + t], false, false);
    }
    a0 = an0;
    a1 = an1;
#pragma unroll
    for (int t = 0; t < 4; ++t) b[t] = bn[t];
  }

  int ro = (lane < 16) ? 0 : 8;                       // C layout: VGPR v -> M = v (+8 hi lanes)
#pragma unroll
  for (int t = 0; t < 4; ++t) {
    float* yb0 = Y + (size_t)(m0 + ro) * H_ + n0 + t * 16 + lm;
    float* yb1 = yb0 + (size_t)16 * H_;
#pragma unroll
    for (int v = 0; v < 8; ++v) {
      yb0[(size_t)v * H_] = c[t][v];
      yb1[(size_t)v * H_] = c[4 + t][v];
    }
  }
}

// ---------------- self-loop init: acc = dinv^2 * h ----------------
__global__ void self_init(const float* __restrict__ h, const float* __restrict__ dinv,
                          float* __restrict__ acc) {
  size_t idx = (size_t)blockIdx.x * blockDim.x + threadIdx.x;  // G*N*H/4 threads
  size_t gn = idx / (H_ / 4);
  int c4 = (int)(idx - gn * (H_ / 4));
  float dv = dinv[gn];
  float nm = dv * dv;
  const float4 v = *(const float4*)(h + gn * H_ + c4 * 4);
  float4 r;
  r.x = nm * v.x; r.y = nm * v.y; r.z = nm * v.z; r.w = nm * v.w;
  *(float4*)(acc + gn * H_ + c4 * 4) = r;
}

// ---------------- edge aggregation: one wave per edge, 8 ch/lane ----------------
__global__ void edge_agg(const int* __restrict__ ei, const float* __restrict__ ew,
                         const float* __restrict__ dinv, const float* __restrict__ h,
                         float* __restrict__ acc) {
  long long tid = (long long)blockIdx.x * blockDim.x + threadIdx.x;  // G*E*32 threads
  int lane = (int)(tid & 31);
  long long we = tid >> 5;
  int g = (int)(we / E_);
  int e = (int)(we - (long long)g * E_);
  const int* base = ei + (size_t)g * 2 * E_;
  int s = base[e];
  int t = base[E_ + e];
  float w = ew[(size_t)g * E_ + e];
  float nm = dinv[g * N_ + s] * w * dinv[g * N_ + t];
  const float* hs = h + ((size_t)g * N_ + s) * H_ + lane * 8;
  float* ap = acc + ((size_t)g * N_ + t) * H_ + lane * 8;
  float4 p0 = *(const float4*)(hs);
  float4 p1 = *(const float4*)(hs + 4);
  atomicAdd(ap + 0, nm * p0.x);
  atomicAdd(ap + 1, nm * p0.y);
  atomicAdd(ap + 2, nm * p0.z);
  atomicAdd(ap + 3, nm * p0.w);
  atomicAdd(ap + 4, nm * p1.x);
  atomicAdd(ap + 5, nm * p1.y);
  atomicAdd(ap + 6, nm * p1.z);
  atomicAdd(ap + 7, nm * p1.w);
}

// ---------------- bias + relu + bf16 convert ----------------
__global__ void bias_relu_bf16(const float* __restrict__ acc, const float* __restrict__ b,
                               unsigned short* __restrict__ xout) {
  size_t idx = (size_t)blockIdx.x * blockDim.x + threadIdx.x;  // G*N*H threads
  int c = (int)(idx & (H_ - 1));
  float v = acc[idx] + b[c];
  v = v > 0.0f ? v : 0.0f;
  xout[idx] = f2bf(v);
}

// ---------------- mean pool over nodes ----------------
__global__ void pool_mean(const unsigned short* __restrict__ xbf, float* __restrict__ pooled) {
  int g = blockIdx.x;
  int c = threadIdx.x;                                // H threads
  const unsigned short* base = xbf + (size_t)g * N_ * H_ + c;
  float s = 0.0f;
  for (int n = 0; n < N_; ++n) s += bf2f(base[(size_t)n * H_]);
  pooled[(size_t)g * H_ + c] = s * (1.0f / N_);
}

// ---------------- final FC head ----------------
__global__ void fc_head(const float* __restrict__ ph, const float* __restrict__ pa,
                        const float* __restrict__ hf, const float* __restrict__ af,
                        const float* __restrict__ fcW, const float* __restrict__ fcb,
                        float* __restrict__ out) {
  int gid = blockIdx.x * blockDim.x + threadIdx.x;    // G*C threads
  if (gid >= G_ * C_) return;
  int g = gid / C_;
  int c = gid - g * C_;
  float s = fcb[c];
  for (int j = 0; j < H_; ++j) s += ph[(size_t)g * H_ + j] * fcW[j * C_ + c];
  for (int j = 0; j < FG_; ++j) s += hf[(size_t)g * FG_ + j] * fcW[(H_ + j) * C_ + c];
  for (int j = 0; j < H_; ++j) s += pa[(size_t)g * H_ + j] * fcW[(H_ + FG_ + j) * C_ + c];
  for (int j = 0; j < FG_; ++j) s += af[(size_t)g * FG_ + j] * fcW[(2 * H_ + FG_ + j) * C_ + c];
  out[gid] = s;
}

// ---------------- host orchestration ----------------
static void run_side(const float* x_in, const int* ei, const float* ew,
                     const float* W0, const unsigned short* Wt, const float* bs,
                     unsigned short* x_bf, float* h, float* acc, float* dinv,
                     float* pooled, hipStream_t stream) {
  const int GN  = G_ * N_;
  const int GE  = G_ * E_;
  const int GNH = G_ * N_ * H_;

  deg_init  <<<GN / 256, 256, 0, stream>>>(dinv);
  deg_accum <<<GE / 256, 256, 0, stream>>>(ei, ew, dinv);
  deg_rsqrt <<<GN / 256, 256, 0, stream>>>(dinv);

  // layer 0 (K = 4, scalar)
  l0_matmul     <<<GNH / 256, 256, 0, stream>>>(x_in, W0, h);
  self_init     <<<(GNH / 4) / 256, 256, 0, stream>>>(h, dinv, acc);
  edge_agg      <<<(size_t)GE * 32 / 256, 256, 0, stream>>>(ei, ew, dinv, h, acc);
  bias_relu_bf16<<<GNH / 256, 256, 0, stream>>>(acc, bs, x_bf);

  // layers 1..4 (K = 256, WMMA); block = 256(M) x 64(N) super-tile
  const int gemm_blocks = (GN / 256) * (H_ / 64);
  for (int l = 1; l < L_; ++l) {
    gemm_bf16_wmma<<<gemm_blocks, 256, 0, stream>>>(
        x_bf, Wt + (size_t)(l - 1) * H_ * H_, h, GN);
    self_init     <<<(GNH / 4) / 256, 256, 0, stream>>>(h, dinv, acc);
    edge_agg      <<<(size_t)GE * 32 / 256, 256, 0, stream>>>(ei, ew, dinv, h, acc);
    bias_relu_bf16<<<GNH / 256, 256, 0, stream>>>(acc, bs + (size_t)l * H_, x_bf);
  }

  pool_mean<<<G_, H_, 0, stream>>>(x_bf, pooled);
}

extern "C" void kernel_launch(void* const* d_in, const int* in_sizes, int n_in,
                              void* d_out, int out_size, void* d_ws, size_t ws_size,
                              hipStream_t stream) {
  (void)in_sizes; (void)n_in; (void)out_size; (void)ws_size;

  const float* home_x = (const float*)d_in[0];
  const float* away_x = (const float*)d_in[1];
  const int*   hei    = (const int*)d_in[2];
  const int*   aei    = (const int*)d_in[3];
  const float* hew    = (const float*)d_in[4];
  const float* aew    = (const float*)d_in[5];
  const float* hf     = (const float*)d_in[6];
  const float* af     = (const float*)d_in[7];
  const float* W0     = (const float*)d_in[8];
  const float* Wh     = (const float*)d_in[9];
  const float* bs     = (const float*)d_in[10];
  const float* fcW    = (const float*)d_in[11];
  const float* fcb    = (const float*)d_in[12];
  float* out = (float*)d_out;

  // workspace partition
  char* wsb = (char*)d_ws;
  size_t off = 0;
  auto alloc = [&](size_t bytes) -> void* {
    void* p = wsb + off;
    off = (off + bytes + 255) & ~(size_t)255;
    return p;
  };
  const size_t GNH = (size_t)G_ * N_ * H_;
  unsigned short* x_bf     = (unsigned short*)alloc(GNH * 2);
  float*          h_buf    = (float*)alloc(GNH * 4);
  float*          acc_buf  = (float*)alloc(GNH * 4);
  float*          dinv_buf = (float*)alloc((size_t)G_ * N_ * 4);
  unsigned short* Wt       = (unsigned short*)alloc((size_t)(L_ - 1) * H_ * H_ * 2);
  float*          pooled_h = (float*)alloc((size_t)G_ * H_ * 4);
  float*          pooled_a = (float*)alloc((size_t)G_ * H_ * 4);

  // convert hidden-layer weights once per launch: Wt[l][n][k] = bf16(Wh[l][k][n])
  wt_conv<<<((L_ - 1) * H_ * H_) / 256, 256, 0, stream>>>(Wh, Wt);

  run_side(home_x, hei, hew, W0, Wt, bs, x_bf, h_buf, acc_buf, dinv_buf, pooled_h, stream);
  run_side(away_x, aei, aew, W0, Wt, bs, x_bf, h_buf, acc_buf, dinv_buf, pooled_a, stream);

  fc_head<<<(G_ * C_ + 255) / 256, 256, 0, stream>>>(pooled_h, pooled_a, hf, af, fcW, fcb, out);
}
